// MaskedContrastiveLoss_1726576855956
// MI455X (gfx1250) — compile-verified, matrix operand order checked by reading, not run on hardware
//
#include <hip/hip_runtime.h>
#include <hip/hip_bf16.h>

// MaskedContrastiveLoss on gfx1250 (MI455X).
// logits = feat_A @ feat_B^T  (256 x 131072 x 256, fp32 inputs)
//
// bf16x3 split GEMM (hi*hi + hi*lo + lo*hi) on V_WMMA_F32_16X16X32_BF16:
// ~fp32 accuracy while staying near the HBM roof (256 MB @ 23.3 TB/s ~ 11us).
// v3: 4x2 register tiling per wave (64x32 output patch). Each K-step converts
// 6 fragments (hardware v_cvt_pk_bf16_f32 + compiler-fused v_fma_mix_f32_bf16,
// ~2 VALU/element = algorithmic floor) and feeds 24 WMMAs -> ~8 VALU/WMMA and
// 0.5 KB fetched per WMMA. 4-wave blocks share the same 64 A-rows in L0.
// Split-K (128 chunks) -> 1024 blocks / 4096 waves; partials accumulate via
// global_atomic_add_f32; a single-block kernel finishes the masked LSE loss.

#define DDIM 131072
#define BDIM 256
#define KCHUNKS 128
#define KCHUNK (DDIM / KCHUNKS)        // 1024
#define INV_TEMP 14.2857142857142857f  // 1/0.07

typedef __attribute__((ext_vector_type(16))) __bf16 v16bf;
typedef __attribute__((ext_vector_type(8)))  float  v8f;
typedef __attribute__((ext_vector_type(8)))  unsigned v8u;

// Hardware pack-convert: D[15:0]=bf16(a), D[31:16]=bf16(b), RNE.
__device__ __forceinline__ unsigned cvt_pk_bf16(float a, float b) {
    unsigned r;
    asm("v_cvt_pk_bf16_f32 %0, %1, %2" : "=v"(r) : "v"(a), "v"(b));
    return r;
}

__device__ __forceinline__ void load16(const float* __restrict__ p, float f[16]) {
    const float4* q = reinterpret_cast<const float4*>(p);
    float4 x0 = q[0], x1 = q[1], x2 = q[4], x3 = q[5];  // K {0..7} and {16..23}
    f[0]=x0.x;  f[1]=x0.y;  f[2]=x0.z;  f[3]=x0.w;
    f[4]=x1.x;  f[5]=x1.y;  f[6]=x1.z;  f[7]=x1.w;
    f[8]=x2.x;  f[9]=x2.y;  f[10]=x2.z; f[11]=x2.w;
    f[12]=x3.x; f[13]=x3.y; f[14]=x3.z; f[15]=x3.w;
}

// fp32 -> bf16 hi + bf16 lo-residual, packed 2-at-a-time.
__device__ __forceinline__ void split16(const float f[16], v8u& hi, v8u& lo) {
    #pragma unroll
    for (int p = 0; p < 8; ++p) {
        float e0 = f[2 * p], e1 = f[2 * p + 1];
        unsigned h = cvt_pk_bf16(e0, e1);
        hi[p] = h;
        float h0 = __builtin_bit_cast(float, h << 16);          // lo-residual:
        float h1 = __builtin_bit_cast(float, h & 0xFFFF0000u);  // fused by clang
        lo[p] = cvt_pk_bf16(e0 - h0, e1 - h1);                  // into fma_mix
    }
}

__device__ __forceinline__ v16bf as_bf(v8u v) {
    return __builtin_bit_cast(v16bf, v);
}

// 3-product accumulate for one 16x16 tile.
__device__ __forceinline__ v8f wmma3(v8u ah, v8u al, v8u bh, v8u bl, v8f c) {
    c = __builtin_amdgcn_wmma_f32_16x16x32_bf16(false, as_bf(ah), false, as_bf(bh),
                                                (short)0, c, false, false);
    c = __builtin_amdgcn_wmma_f32_16x16x32_bf16(false, as_bf(ah), false, as_bf(bl),
                                                (short)0, c, false, false);
    c = __builtin_amdgcn_wmma_f32_16x16x32_bf16(false, as_bf(al), false, as_bf(bh),
                                                (short)0, c, false, false);
    return c;
}

// ---------------------------------------------------------------------------
__global__ void zero_logits_kernel(float* __restrict__ p) {
    int t = blockIdx.x * blockDim.x + threadIdx.x;
    if (t < BDIM * BDIM) p[t] = 0.0f;
}

// ---------------------------------------------------------------------------
// grid (2,4,KCHUNKS), block (32,4). Block covers M=64 x N=128 of C; wave w
// owns the 64x32 patch at colbase + w*32; 4x2 register tiles of 16x16.
// All 4 waves of a block read the same 64 A-rows -> L0 reuse.
// ---------------------------------------------------------------------------
__global__ __launch_bounds__(128)
void gemm_bf16x3_kernel(const float* __restrict__ A,
                        const float* __restrict__ Bm,
                        float* __restrict__ logits) {
    const int lane    = threadIdx.x;                   // 0..31
    const int w       = threadIdx.y;                   // 0..3
    const int rowbase = blockIdx.y * 64;
    const int colbase = blockIdx.x * 128 + w * 32;
    const int kc      = blockIdx.z;

    const int r     = lane & 15;           // row (A) / col (B) within tile
    const int khalf = (lane >> 4) * 8;     // ISA 16-bit operand K sub-block

    const float* __restrict__ arow[4];
    #pragma unroll
    for (int i = 0; i < 4; ++i)
        arow[i] = A + (size_t)(rowbase + i * 16 + r) * DDIM;
    const float* __restrict__ brow[2];
    #pragma unroll
    for (int j = 0; j < 2; ++j)
        brow[j] = Bm + (size_t)(colbase + j * 16 + r) * DDIM;

    v8f acc[4][2];
    #pragma unroll
    for (int i = 0; i < 4; ++i)
        #pragma unroll
        for (int j = 0; j < 2; ++j)
            acc[i][j] = (v8f){};

    const int k0 = kc * KCHUNK;

    for (int kb = k0; kb < k0 + KCHUNK; kb += 32) {
        // Stream ahead (global_prefetch_b8).
        __builtin_prefetch(arow[0] + kb + 256, 0, 1);
        __builtin_prefetch(brow[0] + kb + 256, 0, 1);

        v8u ah[4], al[4], bh[2], bl[2];
        #pragma unroll
        for (int i = 0; i < 4; ++i) {
            float fa[16];
            load16(arow[i] + kb + khalf, fa);
            split16(fa, ah[i], al[i]);
        }
        #pragma unroll
        for (int j = 0; j < 2; ++j) {
            float fb[16];
            load16(brow[j] + kb + khalf, fb);
            split16(fb, bh[j], bl[j]);
        }

        #pragma unroll
        for (int i = 0; i < 4; ++i)
            #pragma unroll
            for (int j = 0; j < 2; ++j)
                acc[i][j] = wmma3(ah[i], al[i], bh[j], bl[j], acc[i][j]);
    }

    // C/D layout: lane(0..15): N=lane, VGPR v -> M=v; lane(16..31): M=v+8.
    const int mhalf = (lane >> 4) ? 8 : 0;
    #pragma unroll
    for (int i = 0; i < 4; ++i) {
        #pragma unroll
        for (int j = 0; j < 2; ++j) {
            float* dst = logits + (size_t)(rowbase + i * 16 + mhalf) * BDIM
                                + colbase + j * 16 + (lane & 15);
            #pragma unroll
            for (int v = 0; v < 8; ++v)
                atomicAdd(dst + (size_t)v * BDIM, acc[i][j][v]);
        }
    }
}

// ---------------------------------------------------------------------------
// Masked log-sum-exp loss; mask (idx[i]!=idx[j])||(i==j) is symmetric so it
// serves both row (a2b) and column (b2a) sums.
// ---------------------------------------------------------------------------
__global__ __launch_bounds__(256)
void loss_reduce_kernel(const float* __restrict__ logits,
                        const int* __restrict__ index,
                        float* __restrict__ out) {
    __shared__ int   s_idx[BDIM];
    __shared__ float s_red[BDIM];
    const int i = threadIdx.x;
    s_idx[i] = index[i];
    __syncthreads();

    const int myidx = s_idx[i];
    const float pos = logits[(size_t)i * BDIM + i] * INV_TEMP;

    float row_sum = 0.0f, col_sum = 0.0f;
    for (int j = 0; j < BDIM; ++j) {
        float m  = ((s_idx[j] != myidx) || (j == i)) ? 1.0f : 0.0f;
        float lr = logits[(size_t)i * BDIM + j] * INV_TEMP;   // row i
        float lc = logits[(size_t)j * BDIM + i] * INV_TEMP;   // col i
        row_sum += m * expf(lr);
        col_sum += m * expf(lc);
    }
    s_red[i] = (-pos + logf(row_sum)) + (-pos + logf(col_sum));
    __syncthreads();

    for (int s = 128; s > 0; s >>= 1) {
        if (i < s) s_red[i] += s_red[i + s];
        __syncthreads();
    }
    if (i == 0) out[0] = s_red[0] * (0.5f / (float)BDIM);
}

// ---------------------------------------------------------------------------
extern "C" void kernel_launch(void* const* d_in, const int* in_sizes, int n_in,
                              void* d_out, int out_size, void* d_ws, size_t ws_size,
                              hipStream_t stream) {
    (void)in_sizes; (void)n_in; (void)out_size; (void)ws_size;
    const float* featA = (const float*)d_in[0];
    const float* featB = (const float*)d_in[1];
    const int*   index = (const int*)d_in[2];
    float*       out    = (float*)d_out;
    float*       logits = (float*)d_ws;           // 256*256*4 = 256 KB

    zero_logits_kernel<<<(BDIM * BDIM + 255) / 256, 256, 0, stream>>>(logits);

    dim3 grid(2, 4, KCHUNKS);    // 2x4 (64x128)-blocks x 128 K-chunks
    dim3 block(32, 4, 1);        // 4 wave32s per block, each owns 64x32
    gemm_bf16x3_kernel<<<grid, block, 0, stream>>>(featA, featB, logits);

    loss_reduce_kernel<<<1, BDIM, 0, stream>>>(logits, index, out);
}
